// BigramNeuralNetwork_48395691492071
// MI455X (gfx1250) — matrix operand order repaired and checked by reading, not compiled, not run on hardware
//
#include <hip/hip_runtime.h>
#include <hip/hip_bf16.h>

// ---- model dims (match reference) ----
#define BV      32000
#define DMODEL  1024
#define NLAYERS 4
#define DINNER  2048
#define DSTATE  16
#define DCONV   4
#define DTRANK  64
#define BATCH   2
#define SEQ     1024
#define BT      (BATCH*SEQ)
#define XPROJ_N (DTRANK + 2*DSTATE)   // 96

typedef __bf16 bf16;
typedef __attribute__((ext_vector_type(16))) bf16  v16bf;
typedef __attribute__((ext_vector_type(8)))  float v8f;

__device__ __forceinline__ v16bf pack16(float4 x0, float4 x1, float4 x2, float4 x3) {
    v16bf v;
    v[0]  = (bf16)x0.x; v[1]  = (bf16)x0.y; v[2]  = (bf16)x0.z; v[3]  = (bf16)x0.w;
    v[4]  = (bf16)x1.x; v[5]  = (bf16)x1.y; v[6]  = (bf16)x1.z; v[7]  = (bf16)x1.w;
    v[8]  = (bf16)x2.x; v[9]  = (bf16)x2.y; v[10] = (bf16)x2.z; v[11] = (bf16)x2.w;
    v[12] = (bf16)x3.x; v[13] = (bf16)x3.y; v[14] = (bf16)x3.z; v[15] = (bf16)x3.w;
    return v;
}

// ---------------- fp32 -> bf16 weight staging ----------------
__global__ __launch_bounds__(256)
void f32_to_bf16_kernel(const float* __restrict__ src, bf16* __restrict__ dst, long n)
{
    long i = (long)blockIdx.x * 256 + threadIdx.x;
    if (i < n) dst[i] = (bf16)src[i];
}

// ----------------------------------------------------------------------------
// GEMM:  C[M,N] = act( A_f32[M,K] @ W_bf16[N,K]^T + bias[N] )  (+= if accum)
// One wave owns a 32 x (16*NB) output tile: 2 A fragments (fp32 loaded,
// converted to bf16 in registers, reused across all NB column fragments) x
// NB B fragments (direct 32B v16bf loads from pre-staged bf16 weights)
// -> 2*NB v_wmma_f32_16x16x32_bf16 per 32-wide K step.
// Requires M%32==0, N%(16*NB)==0, K%32==0. NB=4 for N%64==0 (all big GEMMs),
// NB=2 for the N=96 xproj GEMM. Control flow is wave-uniform, so EXEC is
// all-ones at every WMMA.  act: 0=none, 1=relu, 2=softplus
// ----------------------------------------------------------------------------
template <int NB>
__global__ __launch_bounds__(256)
void gemm_wmma(const float* __restrict__ A, int lda,
               const bf16* __restrict__ W, int ldw,
               const float* __restrict__ bias,
               float* __restrict__ C, int ldc,
               int M, int N, int K, int act, int accum)
{
    const int wave  = (blockIdx.x * blockDim.x + threadIdx.x) >> 5;
    const int lane  = threadIdx.x & 31;
    const int np    = N / (16 * NB);             // column tiles
    const int total = (M >> 5) * np;
    if (wave >= total) return;                   // wave-uniform exit

    const int tm   = wave / np;
    const int tn   = wave % np;
    const int half = lane >> 4;
    const int r    = lane & 15;

    const float* Arow0 = A + (size_t)(tm * 32 + r) * lda;    // tile rows 0..15
    const float* Arow1 = Arow0 + (size_t)16 * lda;           // tile rows 16..31
    const bf16*  Wrow[NB];
    #pragma unroll
    for (int j = 0; j < NB; ++j)
        Wrow[j] = W + (size_t)(tn * 16 * NB + j * 16 + r) * ldw;

    v8f c0[NB], c1[NB];                          // c0: rows 0..15, c1: rows 16..31
    #pragma unroll
    for (int j = 0; j < NB; ++j)
        #pragma unroll
        for (int i = 0; i < 8; ++i) { c0[j][i] = 0.f; c1[j][i] = 0.f; }

    for (int k0 = 0; k0 < K; k0 += 32) {
        // A fragments: lane holds M=r, K runs [k0+16h, +16)
        const float* pa0 = Arow0 + k0 + half * 16;
        const float* pa1 = Arow1 + k0 + half * 16;
        float4 a00 = *(const float4*)(pa0),      a01 = *(const float4*)(pa0 + 4);
        float4 a02 = *(const float4*)(pa0 + 8), a03 = *(const float4*)(pa0 + 12);
        float4 a10 = *(const float4*)(pa1),      a11 = *(const float4*)(pa1 + 4);
        float4 a12 = *(const float4*)(pa1 + 8), a13 = *(const float4*)(pa1 + 12);

        // B fragments: lane holds N=r, 16 contiguous bf16 at K = k0 + 16h (32B aligned)
        v16bf b[NB];
        #pragma unroll
        for (int j = 0; j < NB; ++j) {
            const bf16* pw = Wrow[j] + k0 + half * 16;
            if (k0 + 32 < K)                     // uniform
                __builtin_prefetch(pw + 32, 0, 3);   // -> global_prefetch_b8
            b[j] = *(const v16bf*)(pw);
        }

        v16bf a0 = pack16(a00, a01, a02, a03);
        v16bf a1 = pack16(a10, a11, a12, a13);

        #pragma unroll
        for (int j = 0; j < NB; ++j) {
            c0[j] = __builtin_amdgcn_wmma_f32_16x16x32_bf16(false, a0, false, b[j],
                                                            (short)0, c0[j], false, false);
            c1[j] = __builtin_amdgcn_wmma_f32_16x16x32_bf16(false, a1, false, b[j],
                                                            (short)0, c1[j], false, false);
        }
    }

    // Epilogue. Element i of a 16x16 C tile -> row (8*half + i), col r.
    #pragma unroll
    for (int j = 0; j < NB; ++j) {
        const int col = tn * 16 * NB + j * 16 + r;
        const float bb = bias ? bias[col] : 0.f;
        #pragma unroll
        for (int i = 0; i < 8; ++i) {
            const int row0 = tm * 32 + half * 8 + i;
            const int row1 = row0 + 16;
            float v0 = c0[j][i] + bb;
            float v1 = c1[j][i] + bb;
            if (act == 1) {
                v0 = fmaxf(v0, 0.f); v1 = fmaxf(v1, 0.f);
            } else if (act == 2) {
                v0 = (v0 > 20.f) ? v0 : log1pf(__expf(v0));
                v1 = (v1 > 20.f) ? v1 : log1pf(__expf(v1));
            }
            size_t o0 = (size_t)row0 * ldc + col;
            size_t o1 = (size_t)row1 * ldc + col;
            if (accum) { C[o0] += v0; C[o1] += v1; }
            else       { C[o0]  = v0; C[o1]  = v1; }
        }
    }
}

// ---------------- embedding: x = tok_emb[idx] + pos_emb[t] ----------------
__global__ __launch_bounds__(256)
void embed_kernel(const int* __restrict__ idx, const float* __restrict__ tok,
                  const float* __restrict__ pos, float* __restrict__ x)
{
    int gid = blockIdx.x * 256 + threadIdx.x;
    if (gid >= BT * DMODEL) return;
    int col = gid % DMODEL;
    int bt  = gid / DMODEL;
    int t   = bt % SEQ;
    x[gid] = tok[(size_t)idx[bt] * DMODEL + col] + pos[(size_t)t * DMODEL + col];
}

// ---------------- LayerNorm (one block per row) ----------------
__global__ __launch_bounds__(256)
void layernorm_kernel(const float* __restrict__ x, const float* __restrict__ g,
                      const float* __restrict__ b, float* __restrict__ o)
{
    __shared__ float red[256];
    const int row = blockIdx.x;
    const float* xr = x + (size_t)row * DMODEL;
    float s = 0.f;
    for (int i = threadIdx.x; i < DMODEL; i += 256) s += xr[i];
    red[threadIdx.x] = s; __syncthreads();
    for (int off = 128; off > 0; off >>= 1) {
        if (threadIdx.x < off) red[threadIdx.x] += red[threadIdx.x + off];
        __syncthreads();
    }
    const float mean = red[0] * (1.f / DMODEL);
    __syncthreads();
    float v = 0.f;
    for (int i = threadIdx.x; i < DMODEL; i += 256) {
        float d = xr[i] - mean; v += d * d;
    }
    red[threadIdx.x] = v; __syncthreads();
    for (int off = 128; off > 0; off >>= 1) {
        if (threadIdx.x < off) red[threadIdx.x] += red[threadIdx.x + off];
        __syncthreads();
    }
    const float rstd = rsqrtf(red[0] * (1.f / DMODEL) + 1e-5f);
    float* orow = o + (size_t)row * DMODEL;
    for (int i = threadIdx.x; i < DMODEL; i += 256)
        orow[i] = (xr[i] - mean) * rstd * g[i] + b[i];
}

// -------- depthwise causal conv (width 4) + SiLU; xi = xz[:, :DINNER] --------
__global__ __launch_bounds__(256)
void conv_silu_kernel(const float* __restrict__ xz, const float* __restrict__ cw,
                      const float* __restrict__ cb, float* __restrict__ u)
{
    int gid = blockIdx.x * 256 + threadIdx.x;
    if (gid >= BT * DINNER) return;
    int d  = gid % DINNER;
    int bt = gid / DINNER;
    int t  = bt % SEQ;
    int b  = bt / SEQ;
    float acc = cb[d];
    #pragma unroll
    for (int k = 0; k < DCONV; ++k) {
        int tk = t + k - (DCONV - 1);
        if (tk >= 0)
            acc += xz[(size_t)(b * SEQ + tk) * (2 * DINNER) + d] * cw[d * DCONV + k];
    }
    float sig = 1.f / (1.f + __expf(-acc));
    u[gid] = acc * sig;
}

// -------- selective-scan recurrence; block = 256 channels of one batch --------
__global__ __launch_bounds__(256)
void ssm_scan_kernel(const float* __restrict__ dt, const float* __restrict__ u,
                     const float* __restrict__ xdbl, const float* __restrict__ xz,
                     const float* __restrict__ A_log, const float* __restrict__ D_ssm,
                     float* __restrict__ y)
{
    const int b = blockIdx.x / (DINNER / 256);
    const int d = (blockIdx.x % (DINNER / 256)) * 256 + threadIdx.x;
    __shared__ float sB[DSTATE];
    __shared__ float sC[DSTATE];

    float h[DSTATE], Av[DSTATE];
    #pragma unroll
    for (int s = 0; s < DSTATE; ++s) {
        h[s]  = 0.f;
        Av[s] = -__expf(A_log[(size_t)d * DSTATE + s]);
    }
    const float Dv = D_ssm[d];

    for (int t = 0; t < SEQ; ++t) {
        const size_t row = (size_t)(b * SEQ + t);
        if (threadIdx.x < 2 * DSTATE) {
            float v = xdbl[row * XPROJ_N + DTRANK + threadIdx.x];
            if (threadIdx.x < DSTATE) sB[threadIdx.x] = v;
            else                      sC[threadIdx.x - DSTATE] = v;
        }
        __syncthreads();
        const float dtv = dt[row * DINNER + d];
        const float uv  = u [row * DINNER + d];
        const float zv  = xz[row * (2 * DINNER) + DINNER + d];
        const float dtu = dtv * uv;
        float acc = 0.f;
        #pragma unroll
        for (int s = 0; s < DSTATE; ++s) {
            h[s] = __expf(dtv * Av[s]) * h[s] + dtu * sB[s];
            acc += h[s] * sC[s];
        }
        float yv  = acc + Dv * uv;
        float sig = 1.f / (1.f + __expf(-zv));
        y[row * DINNER + d] = yv * (zv * sig);
        __syncthreads();
    }
}

// ---------------- loss = mean over rows of (logsumexp - logit[target]) -------
__global__ void zero_loss(float* loss) { *loss = 0.f; }

__global__ __launch_bounds__(256)
void loss_kernel(const float* __restrict__ logits, const int* __restrict__ tgt,
                 float* __restrict__ loss)
{
    __shared__ float red[256];
    const int row = blockIdx.x;
    const float* lr = logits + (size_t)row * BV;
    float mx = -1e30f;
    for (int i = threadIdx.x; i < BV; i += 256) mx = fmaxf(mx, lr[i]);
    red[threadIdx.x] = mx; __syncthreads();
    for (int off = 128; off > 0; off >>= 1) {
        if (threadIdx.x < off)
            red[threadIdx.x] = fmaxf(red[threadIdx.x], red[threadIdx.x + off]);
        __syncthreads();
    }
    mx = red[0]; __syncthreads();
    float s = 0.f;
    for (int i = threadIdx.x; i < BV; i += 256) s += __expf(lr[i] - mx);
    red[threadIdx.x] = s; __syncthreads();
    for (int off = 128; off > 0; off >>= 1) {
        if (threadIdx.x < off) red[threadIdx.x] += red[threadIdx.x + off];
        __syncthreads();
    }
    if (threadIdx.x == 0) {
        float lse = mx + __logf(red[0]);
        float lv  = lr[tgt[row]];
        atomicAdd(loss, (lse - lv) * (1.f / (float)BT));
    }
}

// ============================================================================
extern "C" void kernel_launch(void* const* d_in, const int* in_sizes, int n_in,
                              void* d_out, int out_size, void* d_ws, size_t ws_size,
                              hipStream_t stream)
{
    (void)in_sizes; (void)n_in; (void)out_size; (void)ws_size;

    const int*   idx     = (const int*)  d_in[0];
    const int*   targets = (const int*)  d_in[1];
    const float* tok_emb = (const float*)d_in[2];
    const float* pos_emb = (const float*)d_in[3];
    const float* ln1_g   = (const float*)d_in[4];
    const float* ln1_b   = (const float*)d_in[5];
    const float* in_w    = (const float*)d_in[6];
    const float* conv_w  = (const float*)d_in[7];
    const float* conv_b  = (const float*)d_in[8];
    const float* xproj_w = (const float*)d_in[9];
    const float* dt_w    = (const float*)d_in[10];
    const float* dt_b    = (const float*)d_in[11];
    const float* A_log   = (const float*)d_in[12];
    const float* D_ssm   = (const float*)d_in[13];
    const float* out_w   = (const float*)d_in[14];
    const float* ln2_g   = (const float*)d_in[15];
    const float* ln2_b   = (const float*)d_in[16];
    const float* ffn_w1  = (const float*)d_in[17];
    const float* ffn_b1  = (const float*)d_in[18];
    const float* ffn_w2  = (const float*)d_in[19];
    const float* ffn_b2  = (const float*)d_in[20];
    const float* lm_w    = (const float*)d_in[21];
    const float* lm_b    = (const float*)d_in[22];

    // ---- workspace layout ----
    // fp32 activations: ~32.3M floats (~129 MB); bf16 staged weights: ~92.8M
    // elems (~186 MB). Total ~315 MB of d_ws.
    float* ws   = (float*)d_ws;
    float* x    = ws;                                  // BT*DMODEL
    float* xn   = x    + (size_t)BT * DMODEL;          // BT*DMODEL
    float* xz   = xn   + (size_t)BT * DMODEL;          // BT*2*DINNER
    float* u    = xz   + (size_t)BT * 2 * DINNER;      // BT*DINNER
    float* xdbl = u    + (size_t)BT * DINNER;          // BT*XPROJ_N
    float* dt   = xdbl + (size_t)BT * XPROJ_N;         // BT*DINNER
    float* y    = dt   + (size_t)BT * DINNER;          // BT*DINNER
    float* ffn  = y    + (size_t)BT * DINNER;          // BT*4*DMODEL

    const long n_inw  = (long)NLAYERS * 2 * DINNER * DMODEL;
    const long n_xpw  = (long)NLAYERS * XPROJ_N * DINNER;
    const long n_dtw  = (long)NLAYERS * DINNER * DTRANK;
    const long n_outw = (long)NLAYERS * DMODEL * DINNER;
    const long n_f1   = (long)NLAYERS * 4 * DMODEL * DMODEL;
    const long n_f2   = (long)NLAYERS * DMODEL * 4 * DMODEL;
    const long n_lmw  = (long)BV * DMODEL;

    bf16* inw_bf  = (bf16*)(ffn + (size_t)BT * 4 * DMODEL);
    bf16* xpw_bf  = inw_bf  + n_inw;
    bf16* dtw_bf  = xpw_bf  + n_xpw;
    bf16* outw_bf = dtw_bf  + n_dtw;
    bf16* f1_bf   = outw_bf + n_outw;
    bf16* f2_bf   = f1_bf   + n_f1;
    bf16* lmw_bf  = f2_bf   + n_f2;

    float* logits = (float*)d_out;                     // BT*BV
    float* lossp  = logits + (size_t)BT * BV;          // 1

    auto cvt = [&](const float* src, bf16* dst, long n) {
        f32_to_bf16_kernel<<<(int)((n + 255) / 256), 256, 0, stream>>>(src, dst, n);
    };
    auto gemm = [&](const float* A, int lda, const bf16* W, int ldw,
                    const float* bias, float* C, int ldc,
                    int M, int N, int K, int act, int accum) {
        if ((N & 63) == 0) {
            int waves  = (M / 32) * (N / 64);
            int blocks = (waves + 7) / 8;
            gemm_wmma<4><<<blocks, 256, 0, stream>>>(A, lda, W, ldw, bias, C, ldc,
                                                     M, N, K, act, accum);
        } else {                                     // N % 32 == 0 (xproj N=96)
            int waves  = (M / 32) * (N / 32);
            int blocks = (waves + 7) / 8;
            gemm_wmma<2><<<blocks, 256, 0, stream>>>(A, lda, W, ldw, bias, C, ldc,
                                                     M, N, K, act, accum);
        }
    };

    // stage all GEMM weights as bf16 (idempotent, deterministic)
    cvt(in_w,    inw_bf,  n_inw);
    cvt(xproj_w, xpw_bf,  n_xpw);
    cvt(dt_w,    dtw_bf,  n_dtw);
    cvt(out_w,   outw_bf, n_outw);
    cvt(ffn_w1,  f1_bf,   n_f1);
    cvt(ffn_w2,  f2_bf,   n_f2);
    cvt(lm_w,    lmw_bf,  n_lmw);

    // x = tok_emb[idx] + pos_emb
    embed_kernel<<<(BT * DMODEL + 255) / 256, 256, 0, stream>>>(idx, tok_emb, pos_emb, x);

    for (int l = 0; l < NLAYERS; ++l) {
        const bf16*  iw = inw_bf  + (size_t)l * 2 * DINNER * DMODEL;
        const float* cw = conv_w  + (size_t)l * DINNER * DCONV;
        const float* cb = conv_b  + (size_t)l * DINNER;
        const bf16*  xw = xpw_bf  + (size_t)l * XPROJ_N * DINNER;
        const bf16*  dw = dtw_bf  + (size_t)l * DINNER * DTRANK;
        const float* db = dt_b    + (size_t)l * DINNER;
        const float* Al = A_log   + (size_t)l * DINNER * DSTATE;
        const float* Ds = D_ssm   + (size_t)l * DINNER;
        const bf16*  ow = outw_bf + (size_t)l * DMODEL * DINNER;
        const bf16*  f1 = f1_bf   + (size_t)l * 4 * DMODEL * DMODEL;
        const float* b1 = ffn_b1  + (size_t)l * 4 * DMODEL;
        const bf16*  f2 = f2_bf   + (size_t)l * DMODEL * 4 * DMODEL;
        const float* b2 = ffn_b2  + (size_t)l * DMODEL;

        // ---- Mamba block ----
        layernorm_kernel<<<BT, 256, 0, stream>>>(x, ln1_g + l * DMODEL, ln1_b + l * DMODEL, xn);
        // xz = xn @ in_w^T
        gemm(xn, DMODEL, iw, DMODEL, nullptr, xz, 2 * DINNER, BT, 2 * DINNER, DMODEL, 0, 0);
        // u = silu(causal_conv(xi))
        conv_silu_kernel<<<(BT * DINNER + 255) / 256, 256, 0, stream>>>(xz, cw, cb, u);
        // xdbl = u @ xproj^T   (N=96 -> NB=2 path)
        gemm(u, DINNER, xw, DINNER, nullptr, xdbl, XPROJ_N, BT, XPROJ_N, DINNER, 0, 0);
        // dt = softplus(xdbl[:, :64] @ dt_w^T + dt_b)
        gemm(xdbl, XPROJ_N, dw, DTRANK, db, dt, DINNER, BT, DINNER, DTRANK, 2, 0);
        // recurrent scan + D skip + z gate
        ssm_scan_kernel<<<BATCH * (DINNER / 256), 256, 0, stream>>>(dt, u, xdbl, xz, Al, Ds, y);
        // x += y @ out_w^T
        gemm(y, DINNER, ow, DINNER, nullptr, x, DMODEL, BT, DMODEL, DINNER, 0, 1);

        // ---- FFN block ----
        layernorm_kernel<<<BT, 256, 0, stream>>>(x, ln2_g + l * DMODEL, ln2_b + l * DMODEL, xn);
        gemm(xn, DMODEL, f1, DMODEL, b1, ffn, 4 * DMODEL, BT, 4 * DMODEL, DMODEL, 1, 0);
        gemm(ffn, 4 * DMODEL, f2, 4 * DMODEL, b2, x, DMODEL, BT, DMODEL, 4 * DMODEL, 0, 1);
    }

    // logits = x @ lm_w^T + lm_b
    gemm(x, DMODEL, lmw_bf, DMODEL, lm_b, logits, BV, BT, BV, DMODEL, 0, 0);

    // loss
    zero_loss<<<1, 1, 0, stream>>>(lossp);
    loss_kernel<<<BT, 256, 0, stream>>>(logits, targets, lossp);
}
